// TransformerBlock_12060268167488
// MI455X (gfx1250) — matrix operand order repaired and checked by reading, not compiled, not run on hardware
//
#include <hip/hip_runtime.h>

typedef unsigned short u16;
typedef __attribute__((ext_vector_type(16))) __bf16 v16bf;
typedef __attribute__((ext_vector_type(8)))  float  v8f;

// ---------------- CDNA5 feature detection ----------------
#if __has_builtin(__builtin_amdgcn_global_load_async_to_lds_b128)
#define HAVE_ASYNC_COPY 1
#else
#define HAVE_ASYNC_COPY 0
#endif

#if __has_builtin(__builtin_amdgcn_tensor_load_to_lds)
#define HAVE_TDM 1
#else
#define HAVE_TDM 0
#endif

// ---------- helpers ----------
__device__ __forceinline__ u16 f2bf(float f) {
  unsigned u = __float_as_uint(f);
  unsigned r = u + 0x7FFFu + ((u >> 16) & 1u);   // round-to-nearest-even
  return (u16)(r >> 16);
}

// Async global->LDS copy of 16 bytes (falls back to a VGPR copy).
__device__ __forceinline__ void async_copy_b128(const void* gsrc, void* ldst) {
#if HAVE_ASYNC_COPY
  typedef int v4i_ __attribute__((vector_size(16)));
  typedef __attribute__((address_space(1))) v4i_ gv4i;
  typedef __attribute__((address_space(3))) v4i_ lv4i;
  __builtin_amdgcn_global_load_async_to_lds_b128((gv4i*)gsrc, (lv4i*)ldst, 0, 0);
#else
  *(uint4*)ldst = *(const uint4*)gsrc;
#endif
}

__device__ __forceinline__ void wait_async0() {
#if HAVE_ASYNC_COPY
#if __has_builtin(__builtin_amdgcn_s_wait_asynccnt)
  __builtin_amdgcn_s_wait_asynccnt(0);
#else
  asm volatile("s_wait_asynccnt 0x0" ::: "memory");
#endif
#endif
  asm volatile("" ::: "memory");
}

#if HAVE_TDM
typedef unsigned int u32x4 __attribute__((ext_vector_type(4)));
typedef int          i32x4 __attribute__((ext_vector_type(4)));
typedef int          i32x8 __attribute__((ext_vector_type(8)));

__device__ __forceinline__ unsigned lds_off(const void* p) {
  // Generic LDS pointer = {aperture_hi, offset[31:0]}; low dword is the LDS byte offset.
  return (unsigned)(unsigned long long)p;
}

// TDM: 2D dword-tile load, tile = tile_dw dwords x rows, row stride = stride_dw dwords.
// D# bit packing per CDNA5 ISA (count=1, type=2, data_size=4B).
__device__ __forceinline__ void tdm_load_2d_dw(unsigned lds_byte, const void* gsrc,
                                               unsigned tile_dw, unsigned rows,
                                               unsigned stride_dw) {
  unsigned long long ga = (unsigned long long)gsrc;
  u32x4 g0 = { 1u,                                              // count=1 (user, load)
               lds_byte,                                        // lds_addr
               (unsigned)ga,                                    // global_addr[31:0]
               ((unsigned)(ga >> 32) & 0x01FFFFFFu) | 0x80000000u };  // addr[56:32] | type=2
  i32x8 g1;
  g1[0] = 0x00020000;                    // wg_mask=0, data_size=2 (4 bytes)
  g1[1] = (int)(tile_dw << 16);          // tensor_dim0[15:0] (== tile_dim0)
  g1[2] = (int)(rows << 16);             // tensor_dim0[31:16]=0 | tensor_dim1[15:0]
  g1[3] = (int)(tile_dw << 16);          // tensor_dim1[31:16]=0 | tile_dim0
  g1[4] = (int)rows;                     // tile_dim1 | tile_dim2=0
  g1[5] = (int)stride_dw;                // tensor_dim0_stride[31:0]
  g1[6] = 0;                             // stride0 hi | tensor_dim1_stride lo
  g1[7] = 0;                             // tensor_dim1_stride hi
  i32x4 z4 = {0, 0, 0, 0};
#if __clang_major__ >= 23
  i32x8 z8 = {0, 0, 0, 0, 0, 0, 0, 0};
  __builtin_amdgcn_tensor_load_to_lds(g0, g1, z4, z4, z8, 0);
#else
  __builtin_amdgcn_tensor_load_to_lds(g0, g1, z4, z4, 0);
#endif
}

__device__ __forceinline__ void wait_tensor0() {
#if __has_builtin(__builtin_amdgcn_s_wait_tensorcnt)
  __builtin_amdgcn_s_wait_tensorcnt((short)0);
#else
  asm volatile("s_wait_tensorcnt 0x0" ::: "memory");
#endif
  asm volatile("" ::: "memory");
}
#endif  // HAVE_TDM

// A-matrix 16x32 bf16 fragment: lane holds row M=lane&15;
// dword q holds K pair base kb = 2q + 16*half.
__device__ __forceinline__ v16bf load_a_frag(const u16* row, int half) {
  union { v16bf v; unsigned d[8]; } u;
#pragma unroll
  for (int q = 0; q < 8; ++q) {
    int kb = 2 * q + 16 * half;
    u.d[q] = *(const unsigned*)(row + kb);
  }
  return u.v;
}

// B-matrix 32x16 bf16 fragment: lane holds col N=lane&15; lanes 0-15 hold
// K=0..15 (dword q -> K=2q,2q+1), lanes 16-31 hold K=16..31.
// `col` points at K-contiguous storage for that column.
__device__ __forceinline__ v16bf load_b_frag(const u16* col, int half) {
  union { v16bf v; unsigned d[8]; } u;
#pragma unroll
  for (int q = 0; q < 8; ++q) {
    u.d[q] = *(const unsigned*)(col + 2 * q + 16 * half);
  }
  return u.v;
}

// Same B fragment from row-major [K][N] storage with row stride 32 elements.
__device__ __forceinline__ v16bf load_b_frag_strided(const u16* colbase, int half) {
  union { v16bf v; u16 h16[16]; } u;
#pragma unroll
  for (int e = 0; e < 16; ++e) u.h16[e] = colbase[(e + 16 * half) * 32];
  return u.v;
}

// ---------- LayerNorm (one wave per 256-wide row) ----------
__global__ __launch_bounds__(256)
void ln_rows_bf16(const float* __restrict__ x, const float* __restrict__ g,
                  const float* __restrict__ b, u16* __restrict__ out) {
  const int lane = threadIdx.x & 31;
  const int wave = threadIdx.x >> 5;
  const size_t row = (size_t)blockIdx.x * 8 + wave;
  const float* xr = x + row * 256;
  float v[8], s = 0.f, s2 = 0.f;
#pragma unroll
  for (int i = 0; i < 8; ++i) { v[i] = xr[i * 32 + lane]; s += v[i]; s2 += v[i] * v[i]; }
#pragma unroll
  for (int m = 16; m >= 1; m >>= 1) { s += __shfl_xor(s, m, 32); s2 += __shfl_xor(s2, m, 32); }
  float mu   = s * (1.f / 256.f);
  float var  = s2 * (1.f / 256.f) - mu * mu;
  float rstd = rsqrtf(var + 1e-5f);
  u16* orow = out + row * 256;
#pragma unroll
  for (int i = 0; i < 8; ++i) {
    int c = i * 32 + lane;
    orow[c] = f2bf((v[i] - mu) * rstd * g[c] + b[c]);
  }
}

// ---------- Generic bf16 WMMA GEMM: C = act(A*W + bias) (+resid) ----------
// A: [M,K] bf16 row-major (async-staged to LDS).  W: [K,N] fp32 (converted on load).
// Block: 256 thr = 8 waves; tile 64x64; wave -> 16x32 (2 WMMA accumulators).
__global__ __launch_bounds__(256)
void gemm_bf16_wmma(const u16* __restrict__ A, const float* __restrict__ W,
                    const float* __restrict__ bias, const float* __restrict__ resid,
                    float* __restrict__ Cf, u16* __restrict__ Cb,
                    int M, int N, int K, int act) {
  __shared__ __align__(16) u16 aS[64 * 32];      // [m][k]
  __shared__ __align__(16) u16 wS[64 * 34];      // [n][k] transposed, pad->34
  const int tid = threadIdx.x;
  const int lane = tid & 31, wid = tid >> 5;
  const int mw = wid & 3, nw = wid >> 2;
  const int half = lane >> 4, l16 = lane & 15;
  const int bm = blockIdx.y * 64, bn = blockIdx.x * 64;

  v8f acc0 = {0, 0, 0, 0, 0, 0, 0, 0};
  v8f acc1 = {0, 0, 0, 0, 0, 0, 0, 0};

  const int arow = tid >> 2, acol = (tid & 3) * 8;   // A: 64 rows x 32 cols
  const int wrow = tid >> 3, wcol = (tid & 7) * 8;   // W: 32 rows x 64 cols

  for (int kc = 0; kc < K; kc += 32) {
    // A tile: async DMA straight into LDS (no VGPR round-trip)
    async_copy_b128(A + (size_t)(bm + arow) * K + kc + acol, aS + arow * 32 + acol);

    // W tile: fp32 -> bf16 convert, transposed store (overlaps async A copy)
    const float* wp = W + (size_t)(kc + wrow) * N + bn + wcol;
    float4 w0 = *(const float4*)(wp);
    float4 w1 = *(const float4*)(wp + 4);
    if (kc + 32 < K) __builtin_prefetch(wp + (size_t)32 * N, 0, 0);
    float wv[8] = {w0.x, w0.y, w0.z, w0.w, w1.x, w1.y, w1.z, w1.w};
#pragma unroll
    for (int j = 0; j < 8; ++j) wS[(wcol + j) * 34 + wrow] = f2bf(wv[j]);

    wait_async0();
    __syncthreads();

    v16bf a  = load_a_frag(aS + (mw * 16 + l16) * 32, half);
    v16bf b0 = load_b_frag(wS + (nw * 32 + l16) * 34, half);
    v16bf b1 = load_b_frag(wS + (nw * 32 + 16 + l16) * 34, half);
    acc0 = __builtin_amdgcn_wmma_f32_16x16x32_bf16(false, a, false, b0, (short)0, acc0, false, false);
    acc1 = __builtin_amdgcn_wmma_f32_16x16x32_bf16(false, a, false, b1, (short)0, acc1, false, false);
    __syncthreads();
  }

  const int grow0 = bm + mw * 16 + 8 * half;
  const int gc0 = bn + nw * 32 + l16;
  const int gc1 = gc0 + 16;
  const float bi0 = bias ? bias[gc0] : 0.f;
  const float bi1 = bias ? bias[gc1] : 0.f;
#pragma unroll
  for (int r = 0; r < 8; ++r) {
    const size_t grow = (size_t)(grow0 + r);
    float x0 = acc0[r] + bi0;
    float x1 = acc1[r] + bi1;
    if (act == 1) {  // exact GELU
      x0 = 0.5f * x0 * (1.f + erff(x0 * 0.7071067811865475f));
      x1 = 0.5f * x1 * (1.f + erff(x1 * 0.7071067811865475f));
    }
    if (resid) { x0 += resid[grow * N + gc0]; x1 += resid[grow * N + gc1]; }
    if (Cf) { Cf[grow * N + gc0] = x0;       Cf[grow * N + gc1] = x1; }
    if (Cb) { Cb[grow * N + gc0] = f2bf(x0); Cb[grow * N + gc1] = f2bf(x1); }
  }
}

// ---------- Flash attention (bf16 WMMA, online softmax, TDM-staged K/V) ----------
// qkv: [B*S, 768] bf16 (Q @ +0, K @ +256, V @ +512, head h at h*32).
// grid: (S/128, H, B); block 256 = 8 waves; wave owns 16 query rows.
__global__ __launch_bounds__(256)
void flash_attn_bf16(const u16* __restrict__ qkv, u16* __restrict__ o, int S) {
  __shared__ __align__(16) u16 kS[32 * 32];     // [key][d]   (d contiguous)
  __shared__ __align__(16) u16 vS[32 * 32];     // [key][d]   (row-major, strided B gather)
  __shared__ __align__(16) u16 pS[8][16 * 32];  // per-wave P tile [m][key]

  const int tid = threadIdx.x;
  const int lane = tid & 31, wave = tid >> 5;
  const int half = lane >> 4, l16 = lane & 15;
  const int h = blockIdx.y;
  const size_t tokBase = (size_t)blockIdx.z * S;
  const int q0 = blockIdx.x * 128 + wave * 16;

  // Q fragment: 16 rows x 32 d, gathered straight from global (dword aligned)
  v16bf aq;
  {
    union { v16bf v; unsigned d[8]; } u;
    const u16* qrow = qkv + (tokBase + q0 + l16) * 768 + h * 32;
#pragma unroll
    for (int q = 0; q < 8; ++q) {
      int kb = 2 * q + 16 * half;
      u.d[q] = *(const unsigned*)(qrow + kb);
    }
    aq = u.v;
  }

  v8f o0 = {0, 0, 0, 0, 0, 0, 0, 0};
  v8f o1 = {0, 0, 0, 0, 0, 0, 0, 0};
  float mrow[8], lrow[8];
#pragma unroll
  for (int r = 0; r < 8; ++r) { mrow[r] = -1e30f; lrow[r] = 0.f; }

  const float sc = 0.1767766952966369f;  // 1/sqrt(32)

  for (int kc = 0; kc < S; kc += 32) {
    __syncthreads();
#if HAVE_TDM
    // Wave 0 drives the Tensor Data Mover: two 2D tiles (32 keys x 16 dwords,
    // row stride 384 dwords) DMA'd into LDS while other waves idle at barrier.
    if (wave == 0) {
      const u16* ksrc = qkv + (tokBase + kc) * 768 + 256 + h * 32;
      const u16* vsrc = qkv + (tokBase + kc) * 768 + 512 + h * 32;
      tdm_load_2d_dw(lds_off(kS), ksrc, 16u, 32u, 384u);
      tdm_load_2d_dw(lds_off(vS), vsrc, 16u, 32u, 384u);
      wait_tensor0();
    }
#else
    {
      const int kRow = tid >> 3, kCol = (tid & 7) * 4;
      const size_t tok = tokBase + kc + kRow;
      *(uint2*)(kS + kRow * 32 + kCol) =
          *(const uint2*)(qkv + tok * 768 + 256 + h * 32 + kCol);
      *(uint2*)(vS + kRow * 32 + kCol) =
          *(const uint2*)(qkv + tok * 768 + 512 + h * 32 + kCol);
    }
#endif
    __syncthreads();

    v16bf bk0 = load_b_frag(kS + l16 * 32, half);
    v16bf bk1 = load_b_frag(kS + (16 + l16) * 32, half);
    v8f z = {0, 0, 0, 0, 0, 0, 0, 0};
    v8f s0 = __builtin_amdgcn_wmma_f32_16x16x32_bf16(false, aq, false, bk0, (short)0, z, false, false);
    v8f s1 = __builtin_amdgcn_wmma_f32_16x16x32_bf16(false, aq, false, bk1, (short)0, z, false, false);

    // online softmax; row M=r+8*half lives in one VGPR across a 16-lane half
#pragma unroll
    for (int r = 0; r < 8; ++r) {
      float x0 = s0[r] * sc, x1 = s1[r] * sc;
      float mx = fmaxf(x0, x1);
#pragma unroll
      for (int m = 8; m >= 1; m >>= 1) mx = fmaxf(mx, __shfl_xor(mx, m, 32));
      float mn = fmaxf(mrow[r], mx);
      float p0 = __expf(x0 - mn), p1 = __expf(x1 - mn);
      float rs = p0 + p1;
#pragma unroll
      for (int m = 8; m >= 1; m >>= 1) rs += __shfl_xor(rs, m, 32);
      float corr = __expf(mrow[r] - mn);
      lrow[r] = lrow[r] * corr + rs;
      mrow[r] = mn;
      o0[r] *= corr;
      o1[r] *= corr;
      const int mi = r + 8 * half;
      pS[wave][mi * 32 + l16]      = f2bf(p0);   // C-layout -> LDS
      pS[wave][mi * 32 + 16 + l16] = f2bf(p1);
    }

    // same-wave LDS store->load is in-order (DScnt); re-gather as A fragment
    v16bf ap  = load_a_frag(&pS[wave][l16 * 32], half);
    v16bf bv0 = load_b_frag_strided(vS + l16, half);
    v16bf bv1 = load_b_frag_strided(vS + 16 + l16, half);
    o0 = __builtin_amdgcn_wmma_f32_16x16x32_bf16(false, ap, false, bv0, (short)0, o0, false, false);
    o1 = __builtin_amdgcn_wmma_f32_16x16x32_bf16(false, ap, false, bv1, (short)0, o1, false, false);
  }

#pragma unroll
  for (int r = 0; r < 8; ++r) {
    float inv = 1.f / lrow[r];
    size_t row = tokBase + q0 + r + 8 * half;
    o[row * 256 + h * 32 + l16]      = f2bf(o0[r] * inv);
    o[row * 256 + h * 32 + 16 + l16] = f2bf(o1[r] * inv);
  }
}

// ---------- host ----------
extern "C" void kernel_launch(void* const* d_in, const int* in_sizes, int n_in,
                              void* d_out, int out_size, void* d_ws, size_t ws_size,
                              hipStream_t stream) {
  (void)in_sizes; (void)n_in; (void)out_size; (void)ws_size;
  const float* x     = (const float*)d_in[0];
  const float* ln1_g = (const float*)d_in[1];
  const float* ln1_b = (const float*)d_in[2];
  const float* ln2_g = (const float*)d_in[3];
  const float* ln2_b = (const float*)d_in[4];
  const float* w_qkv = (const float*)d_in[5];
  const float* b_qkv = (const float*)d_in[6];
  const float* w_out = (const float*)d_in[7];
  const float* b_out = (const float*)d_in[8];
  const float* w_fc1 = (const float*)d_in[9];
  const float* b_fc1 = (const float*)d_in[10];
  const float* w_fc2 = (const float*)d_in[11];
  const float* b_fc2 = (const float*)d_in[12];
  float* out = (float*)d_out;

  const int Mtok = 8192, E = 256, F = 1024, S = 4096, H = 8, B = 2;

  u16*   xn  = (u16*)d_ws;                       //  4 MB  ln1(x) bf16
  u16*   qkv = xn + (size_t)Mtok * E;            // 12 MB  qkv bf16
  u16*   ob  = qkv + (size_t)Mtok * 3 * E;       //  4 MB  attn out bf16
  float* x1  = (float*)(ob + (size_t)Mtok * E);  //  8 MB  x + attn fp32
  u16*   h2  = (u16*)(x1 + (size_t)Mtok * E);    //  4 MB  ln2(x1) bf16
  u16*   h3  = h2 + (size_t)Mtok * E;            // 16 MB  gelu(fc1) bf16

  dim3 blk(256);
  ln_rows_bf16<<<Mtok / 8, blk, 0, stream>>>(x, ln1_g, ln1_b, xn);
  gemm_bf16_wmma<<<dim3(3 * E / 64, Mtok / 64), blk, 0, stream>>>(
      xn, w_qkv, b_qkv, nullptr, nullptr, qkv, Mtok, 3 * E, E, 0);
  flash_attn_bf16<<<dim3(S / 128, H, B), blk, 0, stream>>>(qkv, ob, S);
  gemm_bf16_wmma<<<dim3(E / 64, Mtok / 64), blk, 0, stream>>>(
      ob, w_out, b_out, x, x1, nullptr, Mtok, E, E, 0);
  ln_rows_bf16<<<Mtok / 8, blk, 0, stream>>>(x1, ln2_g, ln2_b, h2);
  gemm_bf16_wmma<<<dim3(F / 64, Mtok / 64), blk, 0, stream>>>(
      h2, w_fc1, b_fc1, nullptr, nullptr, h3, Mtok, F, E, 1);
  gemm_bf16_wmma<<<dim3(E / 64, Mtok / 64), blk, 0, stream>>>(
      h3, w_fc2, b_fc2, x1, out, nullptr, Mtok, E, F, 0);
}
